// TokenCompression_80771154968641
// MI455X (gfx1250) — matrix operand order repaired and checked by reading, not compiled, not run on hardware
//
#include <hip/hip_runtime.h>
#include <hip/hip_bf16.h>
#include <math.h>

// ---------------------------------------------------------------------------
// TokenCompression for MI455X (gfx1250, wave32, WMMA f32_16x16x32_f16)
//
// B=2, S=4096, H=16, D=128, BLOCK=32, CBLOCK=8, STRIDE=16, HC=2048, nb=255.
// Stacked rows (k then v): M = 16320.
// GEMM1: [16320 x 4096] x [4096 x 2048] + b1 -> exact GELU -> H (f16)
// GEMM2: [16320 x 2048] x [2048 x 1024] + b2 -> d_out (f32)
//
// Wave tile 32x64 (8 x v8f accumulators), block tile 64x256 (8 waves, 2Mx4N).
// ---------------------------------------------------------------------------

typedef __attribute__((ext_vector_type(16))) _Float16 v16h;
typedef __attribute__((ext_vector_type(8)))  _Float16 v8h;
typedef __attribute__((ext_vector_type(8)))  float    v8f;

#define S_LEN 4096
#define NHEAD 16
#define DDIM  128
#define NBLK  255
#define ROWS_PER_BATCH (NBLK * NHEAD)   // 4080
#define ROWS_HALF 8160                  // B * NBLK * NHEAD
#define M_TOTAL   16320
#define K1 4096
#define N1 2048
#define K2 2048
#define N2 1024

// ---------------------------------------------------------------------------
__global__ __launch_bounds__(256) void zero_fill_kernel(float* __restrict__ p, int n)
{
    int i = blockIdx.x * blockDim.x + threadIdx.x;
    if (i < n) p[i] = 0.0f;
}

// ---------------------------------------------------------------------------
// Transpose f32 (rows x cols) -> f16 (cols x rows), LDS-tiled.
// ---------------------------------------------------------------------------
__global__ __launch_bounds__(256) void transpose_f32_to_f16_kernel(
    const float* __restrict__ src, _Float16* __restrict__ dst,
    int rows, int cols)
{
    __shared__ float tile[32][33];
    const int bx = blockIdx.x * 32;
    const int by = blockIdx.y * 32;
    const int tx = threadIdx.x;
    const int ty = threadIdx.y;
    #pragma unroll
    for (int i = 0; i < 32; i += 8) {
        int r = by + ty + i, c = bx + tx;
        tile[ty + i][tx] = (r < rows && c < cols) ? src[(size_t)r * cols + c] : 0.0f;
    }
    __syncthreads();
    #pragma unroll
    for (int i = 0; i < 32; i += 8) {
        int c = bx + ty + i, r = by + tx;
        if (c < cols && r < rows)
            dst[(size_t)c * rows + r] = (_Float16)tile[tx][ty + i];
    }
}

// ---------------------------------------------------------------------------
// GEMM1 fused with gather + pos-encoding + bias + exact GELU.
// posp is branch-free: k rows add real pos, v rows add a zero page from ws.
// ---------------------------------------------------------------------------
__global__ __launch_bounds__(256) void gemm1_gelu_kernel(
    const float* __restrict__ kin, const float* __restrict__ vin,
    const float* __restrict__ pos, const float* __restrict__ zpad,
    const _Float16* __restrict__ W1T,
    const float* __restrict__ b1, _Float16* __restrict__ Hout)
{
    const int lane  = threadIdx.x & 31;
    const int wave  = threadIdx.x >> 5;
    const int lmod  = lane & 15;
    const int lhalf = lane >> 4;
    const int mw = blockIdx.x * 64  + (wave & 1) * 32;
    const int nw = blockIdx.y * 256 + (wave >> 1) * 64;

    // Per-subtile source row decode (uniform per wave except h = lmod).
    const float* tsrc[2];
    const float* posp[2];
    #pragma unroll
    for (int im = 0; im < 2; ++im) {
        const int  r   = mw + im * 16 + lmod;
        const bool isK = (r < ROWS_HALF);
        const int  rr  = isK ? r : (r - ROWS_HALF);
        const int  b   = rr / ROWS_PER_BATCH;
        const int  rem = rr - b * ROWS_PER_BATCH;
        const int  blk = rem >> 4;
        const int  h   = rem & 15;
        tsrc[im] = (isK ? kin : vin)
                 + ((size_t)(b * S_LEN + blk * 16) * NHEAD + h) * DDIM;
        posp[im] = isK ? pos : zpad;
    }

    v8f zero = {};
    v8f acc[2][4];
    #pragma unroll
    for (int im = 0; im < 2; ++im)
        #pragma unroll
        for (int t = 0; t < 4; ++t) acc[im][t] = zero;

    for (int k0 = 0; k0 < K1; k0 += 32) {
        // ---- Two A fragments (rows mw..mw+15 and mw+16..mw+31).
        v16h a[2];
        #pragma unroll
        for (int im = 0; im < 2; ++im) {
            #pragma unroll
            for (int c = 0; c < 2; ++c) {
                const int f = k0 + lhalf * 16 + c * 8;     // feature = p*128+d
                const float* sp = tsrc[im] + (f >> 7) * (NHEAD * DDIM) + (f & 127);
                const float* pp = posp[im] + f;
                float4 x0 = *(const float4*)sp;
                float4 x1 = *(const float4*)(sp + 4);
                float4 p0 = *(const float4*)pp;
                float4 p1 = *(const float4*)(pp + 4);
                x0.x += p0.x; x0.y += p0.y; x0.z += p0.z; x0.w += p0.w;
                x1.x += p1.x; x1.y += p1.y; x1.z += p1.z; x1.w += p1.w;
                const int o = c * 8;
                a[im][o+0]=(_Float16)x0.x; a[im][o+1]=(_Float16)x0.y;
                a[im][o+2]=(_Float16)x0.z; a[im][o+3]=(_Float16)x0.w;
                a[im][o+4]=(_Float16)x1.x; a[im][o+5]=(_Float16)x1.y;
                a[im][o+6]=(_Float16)x1.z; a[im][o+7]=(_Float16)x1.w;
            }
        }
        // ---- 4 B fragments, each reused by both A fragments: 8 WMMAs.
        #pragma unroll
        for (int t = 0; t < 4; ++t) {
            const _Float16* bp =
                W1T + (size_t)(nw + t * 16 + lmod) * K1 + k0 + lhalf * 16;
            v16h bf = *(const v16h*)bp;
            #pragma unroll
            for (int im = 0; im < 2; ++im) {
                acc[im][t] = __builtin_amdgcn_wmma_f32_16x16x32_f16(
                    false, a[im], false, bf, (short)0, acc[im][t], false, false);
            }
        }
    }

    // ---- Epilogue: bias + exact GELU, store f16 hidden.
    #pragma unroll
    for (int im = 0; im < 2; ++im) {
        const int rowb = mw + im * 16 + lhalf * 8;
        #pragma unroll
        for (int t = 0; t < 4; ++t) {
            const int col = nw + t * 16 + lmod;
            const float bias = b1[col];
            #pragma unroll
            for (int j = 0; j < 8; ++j) {
                float x = acc[im][t][j] + bias;
                float g = 0.5f * x * (1.0f + erff(x * 0.7071067811865475f));
                Hout[(size_t)(rowb + j) * N1 + col] = (_Float16)g;
            }
        }
    }
}

// ---------------------------------------------------------------------------
// GEMM2: H(f16) x W2T(f16) + b2 -> f32 into d_out (row-major 16320x1024,
// which is exactly k_compressed flat followed by v_compressed flat).
// ---------------------------------------------------------------------------
__global__ __launch_bounds__(256) void gemm2_kernel(
    const _Float16* __restrict__ Hin, const _Float16* __restrict__ W2T,
    const float* __restrict__ b2, float* __restrict__ out)
{
    const int lane  = threadIdx.x & 31;
    const int wave  = threadIdx.x >> 5;
    const int lmod  = lane & 15;
    const int lhalf = lane >> 4;
    const int mw = blockIdx.x * 64  + (wave & 1) * 32;
    const int nw = blockIdx.y * 256 + (wave >> 1) * 64;

    const _Float16* __restrict__ arow[2] = {
        Hin + (size_t)(mw +      lmod) * K2,
        Hin + (size_t)(mw + 16 + lmod) * K2
    };

    v8f zero = {};
    v8f acc[2][4];
    #pragma unroll
    for (int im = 0; im < 2; ++im)
        #pragma unroll
        for (int t = 0; t < 4; ++t) acc[im][t] = zero;

    for (int k0 = 0; k0 < K2; k0 += 32) {
        v16h a[2];
        #pragma unroll
        for (int im = 0; im < 2; ++im) {
            v8h lo = *(const v8h*)(arow[im] + k0 + lhalf * 16);
            v8h hi = *(const v8h*)(arow[im] + k0 + lhalf * 16 + 8);
            a[im] = __builtin_shufflevector(lo, hi,
                        0,1,2,3,4,5,6,7,8,9,10,11,12,13,14,15);
        }
        #pragma unroll
        for (int t = 0; t < 4; ++t) {
            const _Float16* bp =
                W2T + (size_t)(nw + t * 16 + lmod) * K2 + k0 + lhalf * 16;
            v16h bf = *(const v16h*)bp;
            #pragma unroll
            for (int im = 0; im < 2; ++im) {
                acc[im][t] = __builtin_amdgcn_wmma_f32_16x16x32_f16(
                    false, a[im], false, bf, (short)0, acc[im][t], false, false);
            }
        }
    }

    #pragma unroll
    for (int im = 0; im < 2; ++im) {
        const int rowb = mw + im * 16 + lhalf * 8;
        #pragma unroll
        for (int t = 0; t < 4; ++t) {
            const int col = nw + t * 16 + lmod;
            const float bias = b2[col];
            #pragma unroll
            for (int j = 0; j < 8; ++j) {
                out[(size_t)(rowb + j) * N2 + col] = acc[im][t][j] + bias;
            }
        }
    }
}

// ---------------------------------------------------------------------------
// Launch: zero-page + W transposes -> GEMM1+GELU -> GEMM2 (one stream).
// Workspace: W1T (16 MB) | W2T (4 MB) | H (64 MB) | zero page (16 KB).
// ---------------------------------------------------------------------------
extern "C" void kernel_launch(void* const* d_in, const int* in_sizes, int n_in,
                              void* d_out, int out_size, void* d_ws, size_t ws_size,
                              hipStream_t stream) {
    const float* kin = (const float*)d_in[0];
    const float* vin = (const float*)d_in[1];
    const float* W1  = (const float*)d_in[2];
    const float* b1  = (const float*)d_in[3];
    const float* W2  = (const float*)d_in[4];
    const float* b2  = (const float*)d_in[5];
    const float* pos = (const float*)d_in[6];

    char* ws = (char*)d_ws;
    size_t off = 0;
    _Float16* W1T  = (_Float16*)(ws + off); off += (size_t)K1 * N1 * 2;   // 16 MB
    _Float16* W2T  = (_Float16*)(ws + off); off += (size_t)K2 * N2 * 2;   //  4 MB
    _Float16* Hbuf = (_Float16*)(ws + off); off += (size_t)M_TOTAL * N1 * 2; // 64 MB
    float*    zpad = (float*)   (ws + off); off += (size_t)K1 * 4;        // 16 KB

    zero_fill_kernel<<<dim3((K1 + 255) / 256), 256, 0, stream>>>(zpad, K1);

    dim3 tb(32, 8);
    // W1: 4096 x 2048 -> W1T: 2048 x 4096
    transpose_f32_to_f16_kernel<<<dim3(N1 / 32, K1 / 32), tb, 0, stream>>>(
        W1, W1T, K1, N1);
    // W2: 2048 x 1024 -> W2T: 1024 x 2048
    transpose_f32_to_f16_kernel<<<dim3(N2 / 32, K2 / 32), tb, 0, stream>>>(
        W2, W2T, K2, N2);

    // GEMM1: M blocks 16320/64 = 255, N blocks 2048/256 = 8
    gemm1_gelu_kernel<<<dim3(M_TOTAL / 64, N1 / 256), 256, 0, stream>>>(
        kin, vin, pos, zpad, W1T, b1, Hbuf);

    // GEMM2: M blocks 255, N blocks 1024/256 = 4
    gemm2_kernel<<<dim3(M_TOTAL / 64, N2 / 256), 256, 0, stream>>>(
        Hbuf, W2T, b2, (float*)d_out);
}
